// BasisAffinityGAT_42374147342653
// MI455X (gfx1250) — compile-verified, matrix operand order checked by reading, not run on hardware
//
#include <hip/hip_runtime.h>
#include <hip/hip_bf16.h>
#include <math.h>

// ---------------------------------------------------------------------------
// BasisAffinityGAT on MI455X (gfx1250, wave32, WMMA)
// B=32, N=128, D=512, K=8.  All GEMMs run through v_wmma_f32_16x16x32_bf16
// (f32 accumulate).  LDS staging rows are K-contiguous with stride 72 bf16
// (144 B = 9*16 B: 16B-aligned chunks, conflict-free banks).  Pure-bf16 tile
// staging goes through GLOBAL_LOAD_ASYNC_TO_LDS_B128 (ASYNCcnt).
// ---------------------------------------------------------------------------

#define BDIM 32
#define NDIM 128
#define DDIM 512
#define KB   8
#define LSTR 72   // LDS row stride in bf16 elements

typedef __attribute__((ext_vector_type(16))) __bf16 v16bf;
typedef __attribute__((ext_vector_type(8)))  __bf16 v8bf;
typedef __attribute__((ext_vector_type(4)))  __bf16 v4bf;
typedef __attribute__((ext_vector_type(8)))  float  v8f;
typedef __attribute__((ext_vector_type(4)))  float  v4f;

// 16-bit A/B fragment layout (ISA 7.12.2): lane = row (M or N), h = lane>>4.
// elements 0..7  -> K = 8*h + e ;  elements 8..15 -> K = 16 + 8*h + (e-8)
__device__ __forceinline__ v16bf load_frag(const __bf16* rowp, int h) {
  v8bf lo = *(const v8bf*)(rowp + 8 * h);
  v8bf hi = *(const v8bf*)(rowp + 16 + 8 * h);
  return __builtin_shufflevector(lo, hi, 0, 1, 2, 3, 4, 5, 6, 7,
                                 8, 9, 10, 11, 12, 13, 14, 15);
}

__device__ __forceinline__ v8f wmma_bf16(v16bf a, v16bf b, v8f c) {
  return __builtin_amdgcn_wmma_f32_16x16x32_bf16(false, a, false, b,
                                                 (short)0, c, false, false);
}

__device__ __forceinline__ v4bf pack4(float a, float b, float c, float d) {
  v4bf r;
  r[0] = (__bf16)a; r[1] = (__bf16)b; r[2] = (__bf16)c; r[3] = (__bf16)d;
  return r;
}

// CDNA5 async copy: 16 B global -> LDS per lane, tracked by ASYNCcnt.
// lds_off is the 0-based LDS byte offset (low 32 bits of the flat address).
__device__ __forceinline__ void async_copy16(unsigned lds_off,
                                             const __bf16* gptr) {
  asm volatile("global_load_async_to_lds_b128 %0, %1, off"
               :: "v"(lds_off), "v"(gptr) : "memory");
}
__device__ __forceinline__ void wait_async0() {
  asm volatile("s_wait_asynccnt 0x0" ::: "memory");
}
__device__ __forceinline__ unsigned lds_off_of(const void* p) {
  return (unsigned)(size_t)p;  // flat addr low 32 bits == LDS offset
}

// ---------------------------------------------------------------------------
// Kernel 1: fused = [desc | nv] @ fusion_w^T + fusion_b   (M=4096,N=512,K=1024)
// NT GEMM.  Block: 256 thr (8 waves), tile 32(M) x 64(N), K-chunk 64.
// float4 global loads, packed-bf16 ds_store_b64 staging.
// ---------------------------------------------------------------------------
__global__ __launch_bounds__(256)
void fusion_gemm(const float* __restrict__ desc, const float* __restrict__ nv,
                 const float* __restrict__ fw, const float* __restrict__ fb,
                 __bf16* __restrict__ fused) {
  __shared__ __align__(16) __bf16 Als[32 * LSTR];
  __shared__ __align__(16) __bf16 Bls[64 * LSTR];
  const int tid = threadIdx.x;
  const int lane = tid & 31, wave = tid >> 5;
  const int h = lane >> 4, r = lane & 15;
  const int wm = wave >> 2, wn = wave & 3;  // 2 x 4 wave grid
  const int m0 = blockIdx.x * 32;           // gridDim.x = 128
  const int n0 = blockIdx.y * 64;           // gridDim.y = 8

  v8f acc = {};
  for (int kc = 0; kc < 2 * DDIM; kc += 64) {
    // stage A (32 rows x 64 cols = 512 float4 slots, 2 per thread)
#pragma unroll
    for (int t = 0; t < 2; ++t) {
      int idx = t * 256 + tid;
      int row = idx >> 4, c4 = (idx & 15) * 4;
      int gk = kc + c4;  // 64-col chunk never straddles the desc/nv split
      const float* src = (gk < DDIM)
                             ? (desc + (size_t)(m0 + row) * DDIM + gk)
                             : (nv + (size_t)(m0 + row) * DDIM + (gk - DDIM));
      v4f w = *(const v4f*)src;
      *(v4bf*)(&Als[row * LSTR + c4]) = pack4(w[0], w[1], w[2], w[3]);
    }
    // stage B (64 x 64 = 1024 float4 slots, 4 per thread)
#pragma unroll
    for (int t = 0; t < 4; ++t) {
      int idx = t * 256 + tid;
      int row = idx >> 4, c4 = (idx & 15) * 4;
      v4f w = *(const v4f*)(fw + (size_t)(n0 + row) * (2 * DDIM) + kc + c4);
      *(v4bf*)(&Bls[row * LSTR + c4]) = pack4(w[0], w[1], w[2], w[3]);
    }
    __syncthreads();
    const __bf16* arow = &Als[(wm * 16 + r) * LSTR];
    const __bf16* brow = &Bls[(wn * 16 + r) * LSTR];
    acc = wmma_bf16(load_frag(arow, h), load_frag(brow, h), acc);
    acc = wmma_bf16(load_frag(arow + 32, h), load_frag(brow + 32, h), acc);
    __syncthreads();
  }
  const int col = n0 + wn * 16 + r;
  const float bias = fb[col];
  const int mbase = m0 + wm * 16 + 8 * h;
#pragma unroll
  for (int i = 0; i < 8; ++i)
    fused[(size_t)(mbase + i) * DDIM + col] = (__bf16)(acc[i] + bias);
}

// ---------------------------------------------------------------------------
// Kernel 2: q/k = fused[b] @ W_{q,k}[kb]   (per (b,kb): 128 x 512 x 512, NN)
// A tile: async b128 copy (bf16->bf16).  W tile: register-transposed 4x4
// micro-tiles (coalesced b128 reads along f, ds_store_b64 writes along d).
// grid = (32, 256, 2): x = mt*8+ft, y = b*8+kb, z selects q vs k.
// ---------------------------------------------------------------------------
__global__ __launch_bounds__(256)
void proj_gemm(const __bf16* __restrict__ fused, const float* __restrict__ Wq,
               const float* __restrict__ Wk, __bf16* __restrict__ qout,
               __bf16* __restrict__ kout) {
  __shared__ __align__(16) __bf16 Als[32 * LSTR];
  __shared__ __align__(16) __bf16 Bls[64 * LSTR];
  const int tid = threadIdx.x;
  const int lane = tid & 31, wave = tid >> 5;
  const int h = lane >> 4, r = lane & 15;
  const int wm = wave >> 2, wn = wave & 3;
  const int mt = blockIdx.x >> 3;  // 0..3  (128 rows / 32)
  const int ft = blockIdx.x & 7;   // 0..7  (512 cols / 64)
  const int pair = blockIdx.y;     // b*8 + kb
  const int b = pair >> 3, kb = pair & 7;
  const float* W = blockIdx.z ? Wk : Wq;
  __bf16* out = blockIdx.z ? kout : qout;
  const __bf16* Abase = fused + ((size_t)b * NDIM + mt * 32) * DDIM;
  const float* Wb = W + (size_t)kb * DDIM * DDIM;

  // A staging slots: 32 rows x 8 chunks = 256 -> one async per thread
  const int arow_s = tid >> 3, ac8 = (tid & 7) * 8;
  const unsigned a_lds = lds_off_of(&Als[arow_s * LSTR + ac8]);
  // B micro-tile coords: f along lanes (coalesced reads)
  const int dm = (tid >> 4) * 4, fm = (tid & 15) * 4;

  v8f acc = {};
  for (int kc = 0; kc < DDIM; kc += 64) {
    async_copy16(a_lds, Abase + (size_t)arow_s * DDIM + kc + ac8);
    // transpose W[d][f] -> Bls[f][d] through registers
    v4f w0 = *(const v4f*)(Wb + (size_t)(kc + dm + 0) * DDIM + ft * 64 + fm);
    v4f w1 = *(const v4f*)(Wb + (size_t)(kc + dm + 1) * DDIM + ft * 64 + fm);
    v4f w2 = *(const v4f*)(Wb + (size_t)(kc + dm + 2) * DDIM + ft * 64 + fm);
    v4f w3 = *(const v4f*)(Wb + (size_t)(kc + dm + 3) * DDIM + ft * 64 + fm);
#pragma unroll
    for (int i = 0; i < 4; ++i)
      *(v4bf*)(&Bls[(fm + i) * LSTR + dm]) = pack4(w0[i], w1[i], w2[i], w3[i]);
    wait_async0();
    __syncthreads();
    const __bf16* arow = &Als[(wm * 16 + r) * LSTR];
    const __bf16* brow = &Bls[(wn * 16 + r) * LSTR];
    acc = wmma_bf16(load_frag(arow, h), load_frag(brow, h), acc);
    acc = wmma_bf16(load_frag(arow + 32, h), load_frag(brow + 32, h), acc);
    __syncthreads();
  }
  const int col = ft * 64 + wn * 16 + r;
  const int rowbase = mt * 32 + wm * 16 + 8 * h;
  __bf16* obase = out + (size_t)pair * NDIM * DDIM;
#pragma unroll
  for (int i = 0; i < 8; ++i)
    obase[(size_t)(rowbase + i) * DDIM + col] = (__bf16)acc[i];
}

// ---------------------------------------------------------------------------
// Kernel 3: inverse L2 norms.  One wave per 512-element row (q rows then k).
// ---------------------------------------------------------------------------
__global__ __launch_bounds__(256)
void inv_norms(const __bf16* __restrict__ qk, float* __restrict__ inv) {
  const int lane = threadIdx.x & 31;
  const int wave = threadIdx.x >> 5;
  const long row = (long)blockIdx.x * 8 + wave;
  const v8bf* p = (const v8bf*)(qk + row * DDIM);  // 64 chunks of 8
  float s = 0.f;
#pragma unroll
  for (int j = 0; j < 2; ++j) {
    v8bf x = p[j * 32 + lane];
#pragma unroll
    for (int i = 0; i < 8; ++i) {
      float v = (float)x[i];
      s += v * v;
    }
  }
#pragma unroll
  for (int off = 16; off; off >>= 1) s += __shfl_xor(s, off, 32);
  if (lane == 0) inv[row] = 1.0f / fmaxf(sqrtf(s), 1e-12f);
}

// ---------------------------------------------------------------------------
// Kernel 4: logits = (q @ k^T) * qinv*kinv/sqrt(D); row softmax -> alpha.
// One block per (b,kb); each wave owns 16 rows x 128 cols (8 accum tiles).
// Q/K tiles staged with async b128 copies; softmax reduces in 16-lane halves.
// ---------------------------------------------------------------------------
__global__ __launch_bounds__(256)
void attn_softmax(const __bf16* __restrict__ qbf, const __bf16* __restrict__ kbf,
                  const float* __restrict__ inv, float* __restrict__ alpha) {
  __shared__ __align__(16) __bf16 Qls[128 * LSTR];  // 18 KB
  __shared__ __align__(16) __bf16 Kls[128 * LSTR];  // 18 KB
  const int tid = threadIdx.x;
  const int lane = tid & 31, wave = tid >> 5;
  const int h = lane >> 4, r = lane & 15;
  const int pair = blockIdx.x;  // b*8 + kb
  const __bf16* Q = qbf + (size_t)pair * NDIM * DDIM;
  const __bf16* Kt = kbf + (size_t)pair * NDIM * DDIM;

  v8f acc[8] = {};
  for (int kc = 0; kc < DDIM; kc += 64) {
    // 128 rows x 8 chunks per array = 1024 slots -> 4 asyncs per thread each
#pragma unroll
    for (int t = 0; t < 4; ++t) {
      int idx = t * 256 + tid;
      int row = idx >> 3, c8 = (idx & 7) * 8;
      size_t goff = (size_t)row * DDIM + kc + c8;
      unsigned loff = (unsigned)(row * LSTR + c8) * 2u;
      async_copy16(lds_off_of(Qls) + loff, Q + goff);
      async_copy16(lds_off_of(Kls) + loff, Kt + goff);
    }
    wait_async0();
    __syncthreads();
#pragma unroll
    for (int ks = 0; ks < 64; ks += 32) {
      v16bf a = load_frag(&Qls[(wave * 16 + r) * LSTR + ks], h);
#pragma unroll
      for (int mt = 0; mt < 8; ++mt) {
        v16bf bf = load_frag(&Kls[(mt * 16 + r) * LSTR + ks], h);
        acc[mt] = wmma_bf16(a, bf, acc[mt]);
      }
    }
    __syncthreads();
  }

  const float scale = 0.044194173824159216f;  // 1/sqrt(512)
  const float* qinv = inv + (size_t)NDIM * pair;
  const float* kinv = inv + (size_t)NDIM * (256 + pair);

  float qi[8], rmax[8], rsum[8];
#pragma unroll
  for (int i = 0; i < 8; ++i) {
    qi[i] = qinv[wave * 16 + 8 * h + i] * scale;
    rmax[i] = -3.4e38f;
    rsum[i] = 0.f;
  }
#pragma unroll
  for (int mt = 0; mt < 8; ++mt) {
    float ki = kinv[mt * 16 + r];
#pragma unroll
    for (int i = 0; i < 8; ++i) {
      float v = acc[mt][i] * qi[i] * ki;
      acc[mt][i] = v;
      rmax[i] = fmaxf(rmax[i], v);
    }
  }
#pragma unroll
  for (int i = 0; i < 8; ++i)
#pragma unroll
    for (int off = 1; off < 16; off <<= 1)
      rmax[i] = fmaxf(rmax[i], __shfl_xor(rmax[i], off, 32));
#pragma unroll
  for (int mt = 0; mt < 8; ++mt)
#pragma unroll
    for (int i = 0; i < 8; ++i) {
      float e = __expf(acc[mt][i] - rmax[i]);
      acc[mt][i] = e;
      rsum[i] += e;
    }
#pragma unroll
  for (int i = 0; i < 8; ++i) {
#pragma unroll
    for (int off = 1; off < 16; off <<= 1) rsum[i] += __shfl_xor(rsum[i], off, 32);
    rsum[i] = 1.0f / rsum[i];
  }
  float* out = alpha + (size_t)pair * NDIM * NDIM;
#pragma unroll
  for (int mt = 0; mt < 8; ++mt)
#pragma unroll
    for (int i = 0; i < 8; ++i)
      out[(size_t)(wave * 16 + 8 * h + i) * NDIM + mt * 16 + r] =
          acc[mt][i] * rsum[i];
}

// ---------------------------------------------------------------------------
// Kernel 5: bias_log = log(max(0.01 * mean_b(alpha), 1e-6)), broadcast over B.
// float4 loads/stores; one thread per 4 (kb,n,m) cells.
// ---------------------------------------------------------------------------
__global__ __launch_bounds__(256)
void ema_bias(const float* __restrict__ alpha, float* __restrict__ bias) {
  const size_t e0 = ((size_t)blockIdx.x * 256 + threadIdx.x) * 4;
  v4f s = {};
#pragma unroll 8
  for (int b = 0; b < BDIM; ++b)
    s += *(const v4f*)(alpha + (size_t)b * (KB * NDIM * NDIM) + e0);
  v4f r;
#pragma unroll
  for (int i = 0; i < 4; ++i)
    r[i] = logf(fmaxf(s[i] * (0.01f / 32.0f), 1e-6f));
  for (int b = 0; b < BDIM; ++b)
    *(v4f*)(bias + (size_t)b * (KB * NDIM * NDIM) + e0) = r;
}

// ---------------------------------------------------------------------------
extern "C" void kernel_launch(void* const* d_in, const int* in_sizes, int n_in,
                              void* d_out, int out_size, void* d_ws,
                              size_t ws_size, hipStream_t stream) {
  const float* desc = (const float*)d_in[0];
  const float* nv = (const float*)d_in[1];
  const float* Wq = (const float*)d_in[2];
  const float* Wk = (const float*)d_in[3];
  const float* fw = (const float*)d_in[4];
  const float* fb = (const float*)d_in[5];

  float* out = (float*)d_out;
  float* bias = out;                                     // [B,K,N,N]
  float* alpha = out + (size_t)BDIM * KB * NDIM * NDIM;  // [B,K,N,N]

  char* ws = (char*)d_ws;
  __bf16* fused = (__bf16*)ws;                             //  4 MB
  __bf16* qbf = (__bf16*)(ws + (size_t)4 * 1024 * 1024);   // 32 MB
  __bf16* kbf = (__bf16*)(ws + (size_t)36 * 1024 * 1024);  // 32 MB (contig)
  float* inv = (float*)(ws + (size_t)68 * 1024 * 1024);    // 256 KB

  fusion_gemm<<<dim3(128, 8), 256, 0, stream>>>(desc, nv, fw, fb, fused);
  proj_gemm<<<dim3(32, 256, 2), 256, 0, stream>>>(fused, Wq, Wk, qbf, kbf);
  inv_norms<<<(2 * 256 * NDIM) / 8, 256, 0, stream>>>(qbf, inv);
  attn_softmax<<<256, 256, 0, stream>>>(qbf, kbf, inv, alpha);
  ema_bias<<<(KB * NDIM * NDIM) / (256 * 4), 256, 0, stream>>>(alpha, bias);
}
